// ResNet_3161095930026
// MI455X (gfx1250) — matrix operand order repaired and checked by reading, not compiled, not run on hardware
//
#include <hip/hip_runtime.h>

typedef __attribute__((ext_vector_type(16))) _Float16 v16h;
typedef __attribute__((ext_vector_type(8)))  _Float16 v8h;
typedef __attribute__((ext_vector_type(8)))  float    v8f;

#define IN_DIM     201
#define T_STEPS    256
#define LAST_START 55   // T - IN_DIM

// ---- packed f16 weight slots (offsets in halfs), layout [k16][n][j<16] ----
#define OFF_L1    0        // Kpad=224, Npad=64  -> 14336
#define OFF_R1W1  14336    // 64x64              -> 4096
#define OFF_R1W2  18432    // 64x64              -> 4096
#define OFF_R2W1  22528    // Kpad=64, Npad=128  -> 8192
#define OFF_R2W2  30720    // 128x128            -> 16384
#define OFF_R2DW  47104    // Kpad=64, Npad=128  -> 8192
#define OFF_TAIL  55296    // Kpad=128, Npad=16  -> 2048
#define WPACK_HALFS 57344

// ---- bias block (offsets in floats) ----
#define BO_L1    0
#define BO_R1B1  64
#define BO_R1B2  128
#define BO_R2B1  192
#define BO_R2B2  320
#define BO_R2DB  448
#define BO_TAIL  576
#define BIAS_FLOATS 592

// ---- d_ws byte layout ----
#define WS_WPACK 0
#define WS_BIAS  114688
#define WS_WC    117056   // 128*128 f32 temp (wo@wv)
#define WS_BVC   182592   // 128 f32 temp (wo@bv+bo)

// ---- LDS byte layout (dynamic) ----
#define L_WP     0        // packed weights f16           114688 B
#define L_BIAS   114688   // biases f32                     2368 B
#define L_V      117056   // LIF states f32 (64 rows)     114688 B
#define L_XC     231744   // x cache f16, 64x288 (padded)  36864 B
#define L_BUF    268608   // per-wave bufH/bufA f16        32768 B
#define LDS_BYTES 301376

#define ROWS_PER_WG 64
#define XSTRIDE 288       // 256 cols of x + 32 zero pad (guard-free A gather)

#if defined(__HIP_DEVICE_COMPILE__) && \
    __has_builtin(__builtin_amdgcn_global_load_async_to_lds_b128) && \
    __has_builtin(__builtin_amdgcn_s_wait_asynccnt)
#define USE_ASYNC_LDS 1
#else
#define USE_ASYNC_LDS 0
#endif

// ======================= prep kernels (weight packing + tail folding) =======

__device__ __forceinline__ void pack_w(_Float16* dst, const float* src,
                                       int Nout, int Kin, int Npad, int Kpad,
                                       int tid, int nth) {
  const int total = Kpad * Npad;
  for (int idx = tid; idx < total; idx += nth) {
    int k16 = idx / (Npad * 16);
    int rem = idx - k16 * (Npad * 16);
    int n = rem >> 4;
    int j = rem & 15;
    int k = k16 * 16 + j;
    float v = (n < Nout && k < Kin) ? src[n * Kin + k] : 0.0f;
    dst[idx] = (_Float16)v;
  }
}

__global__ void snn_prep1(const float* l1w, const float* l1b,
                          const float* r1w1, const float* r1b1,
                          const float* r1w2, const float* r1b2,
                          const float* r2w1, const float* r2b1,
                          const float* r2w2, const float* r2b2,
                          const float* r2dw, const float* r2db,
                          const float* wv, const float* bv,
                          const float* wo, const float* bo,
                          char* ws) {
  const int tid = blockIdx.x * blockDim.x + threadIdx.x;
  const int nth = gridDim.x * blockDim.x;
  _Float16* wp = (_Float16*)(ws + WS_WPACK);
  float* wb  = (float*)(ws + WS_BIAS);
  float* Wc  = (float*)(ws + WS_WC);
  float* bvc = (float*)(ws + WS_BVC);

  pack_w(wp + OFF_L1,   l1w,  64,  IN_DIM, 64,  224, tid, nth);
  pack_w(wp + OFF_R1W1, r1w1, 64,  64,     64,  64,  tid, nth);
  pack_w(wp + OFF_R1W2, r1w2, 64,  64,     64,  64,  tid, nth);
  pack_w(wp + OFF_R2W1, r2w1, 128, 64,     128, 64,  tid, nth);
  pack_w(wp + OFF_R2W2, r2w2, 128, 128,    128, 128, tid, nth);
  pack_w(wp + OFF_R2DW, r2dw, 128, 64,     128, 64,  tid, nth);

  for (int i = tid; i < 64; i += nth)  wb[BO_L1 + i]   = l1b[i];
  for (int i = tid; i < 64; i += nth)  wb[BO_R1B1 + i] = r1b1[i];
  for (int i = tid; i < 64; i += nth)  wb[BO_R1B2 + i] = r1b2[i];
  for (int i = tid; i < 128; i += nth) wb[BO_R2B1 + i] = r2b1[i];
  for (int i = tid; i < 128; i += nth) wb[BO_R2B2 + i] = r2b2[i];
  for (int i = tid; i < 128; i += nth) wb[BO_R2DB + i] = r2db[i];

  // Wc = wo @ wv  (attention with seq_len=1 collapses to one affine map)
  for (int idx = tid; idx < 128 * 128; idx += nth) {
    int i = idx >> 7, k = idx & 127;
    float s = 0.0f;
    for (int j = 0; j < 128; ++j) s += wo[i * 128 + j] * wv[j * 128 + k];
    Wc[idx] = s;
  }
  // bvc = wo @ bv + bo
  for (int i = tid; i < 128; i += nth) {
    float s = 0.0f;
    for (int j = 0; j < 128; ++j) s += wo[i * 128 + j] * bv[j];
    bvc[i] = s + bo[i];
  }
}

__global__ void snn_prep2(const float* fcw, const float* fcb, char* ws) {
  const int tid = blockIdx.x * blockDim.x + threadIdx.x;
  const int nth = gridDim.x * blockDim.x;
  _Float16* wp = (_Float16*)(ws + WS_WPACK);
  float* wb = (float*)(ws + WS_BIAS);
  const float* Wc  = (const float*)(ws + WS_WC);
  const float* bvc = (const float*)(ws + WS_BVC);

  // tail: Wt[n][k] = sum_q fcw[n][q] * Wc[q][k]   (Nout=10, K=128) packed
  for (int idx = tid; idx < 128 * 16; idx += nth) {
    int k16 = idx / 256;
    int rem = idx - k16 * 256;
    int n = rem >> 4, j = rem & 15;
    int k = k16 * 16 + j;
    float v = 0.0f;
    if (n < 10) {
      for (int q = 0; q < 128; ++q) v += fcw[n * 128 + q] * Wc[q * 128 + k];
    }
    wp[OFF_TAIL + idx] = (_Float16)v;
  }
  for (int n = tid; n < 16; n += nth) {
    float s = 0.0f;
    if (n < 10) {
      for (int k = 0; k < 128; ++k) s += fcw[n * 128 + k] * bvc[k];
      s += fcb[n];
    }
    wb[BO_TAIL + n] = s;
  }
}

// ======================= main persistent scan kernel ========================

__device__ __forceinline__ v8f wmma_f16(v16h a, v16h b, v8f c) {
  return __builtin_amdgcn_wmma_f32_16x16x32_f16(false, a, false, b,
                                                (short)0, c, false, false);
}

// B fragment: packed layout [k16][n][16 contiguous halfs]  -> one 32B LDS read
__device__ __forceinline__ v16h load_b_frag(const _Float16* w, int Npad,
                                            int kt, int nt, int lane) {
  int off = ((kt * 2 + (lane >> 4)) * Npad + nt * 16 + (lane & 15)) * 16;
  return *(const v16h*)(w + off);
}

// A fragment from row-major f16 buffer (stride 128):
// lane L: m=L%16; elems 0..15 -> K = kt*32 + (L/16)*16 + j
__device__ __forceinline__ v16h load_a_frag(const _Float16* buf, int kt, int lane) {
  int base = (lane & 15) * 128 + kt * 32 + (lane >> 4) * 16;
  v8h lo = *(const v8h*)(buf + base);
  v8h hh = *(const v8h*)(buf + base + 8);
  return __builtin_shufflevector(lo, hh, 0, 1, 2, 3, 4, 5, 6, 7,
                                 8, 9, 10, 11, 12, 13, 14, 15);
}

// layer1 A fragment straight from the padded x cache. No bounds check needed:
// packed l1 weights are zero for k>=201, and all reachable cache bytes
// (idx <= start+223 <= 278 < 288) are initialized finite values.
__device__ __forceinline__ v16h load_a_x(const _Float16* xr, int start,
                                         int kt, int lane) {
  const _Float16* row = xr + (lane & 15) * XSTRIDE + start;
  const int kb = kt * 32 + (lane >> 4) * 16;
  v16h a;
#pragma unroll
  for (int j = 0; j < 8; ++j) a[j] = row[kb + j];
#pragma unroll
  for (int j = 0; j < 8; ++j) a[8 + j] = row[kb + 8 + j];
  return a;
}

__device__ __forceinline__ float lif(float* vp, float cur) {
  float v = *vp;
  v += (cur - v) * 0.5f;                  // tau = 2
  float s = (v >= 1.0f) ? 1.0f : 0.0f;    // threshold 1, hard reset
  *vp = v * (1.0f - s);
  return s;
}

template <int KT, int NT>
__device__ __forceinline__ void gemm_buf(v8f* acc, const _Float16* A,
                                         const _Float16* W, int Npad, int lane) {
#pragma unroll
  for (int nt = 0; nt < NT; ++nt)
#pragma unroll
    for (int e = 0; e < 8; ++e) acc[nt][e] = 0.0f;
#pragma unroll
  for (int kt = 0; kt < KT; ++kt) {
    v16h a = load_a_frag(A, kt, lane);
#pragma unroll
    for (int nt = 0; nt < NT; ++nt) {
      v16h b = load_b_frag(W, Npad, kt, nt, lane);
      acc[nt] = wmma_f16(a, b, acc[nt]);
    }
  }
}

__global__ void __launch_bounds__(128)
snn_main(const float* __restrict__ x, const char* __restrict__ ws,
         float* __restrict__ out) {
  extern __shared__ __align__(16) char smem[];
  _Float16* sW  = (_Float16*)(smem + L_WP);
  float*    sB  = (float*)(smem + L_BIAS);
  float*    sV  = (float*)(smem + L_V);
  _Float16* sX  = (_Float16*)(smem + L_XC);
  _Float16* sBf = (_Float16*)(smem + L_BUF);

  const int tid   = threadIdx.x;
  const int lane  = tid & 31;
  const int wave  = tid >> 5;
  const int rowWG = blockIdx.x * ROWS_PER_WG;

  // --- one-time setup: weights+biases (async DMA to LDS), state=0, x cache ---
#if USE_ASYNC_LDS
  {
    // builtin signature (from probe): (int4 AS1* src, int4 AS3* dst, imm, imm)
    typedef __attribute__((__vector_size__(16))) int aint4;
    typedef __attribute__((address_space(1))) aint4 g_aint4;
    typedef __attribute__((address_space(3))) aint4 l_aint4;
    g_aint4* gsrc = (g_aint4*)(unsigned long long)(ws + WS_WPACK);
    l_aint4* ldst = (l_aint4*)(unsigned int)(unsigned long long)(smem + L_WP);
    // weights (114688 B) + biases (2368 B) are contiguous: 7316 x 16B
    for (int i = tid; i < 7316; i += 128)
      __builtin_amdgcn_global_load_async_to_lds_b128(gsrc + i, ldst + i, 0, 0);
    __builtin_amdgcn_s_wait_asynccnt(0);
  }
#else
  {
    const uint4* src = (const uint4*)(ws + WS_WPACK);
    uint4* dst = (uint4*)sW;
    for (int i = tid; i < 7316; i += 128) dst[i] = src[i];
  }
#endif
  for (int i = tid; i < 64 * 64 * 3 + 64 * 128 * 2; i += 128) sV[i] = 0.0f;
  for (int i = tid; i < ROWS_PER_WG * XSTRIDE; i += 128) {
    int r = i / XSTRIDE, c = i - r * XSTRIDE;
    sX[i] = (c < 256) ? (_Float16)x[(rowWG + r) * 256 + c] : (_Float16)0.0f;
  }
  __syncthreads();

  // per-wave views (each wave owns 16 batch rows end-to-end: no sync in loop)
  const _Float16* xw = sX + wave * 16 * XSTRIDE;
  _Float16* bufH = sBf + wave * 4096;   // 16x128 f16
  _Float16* bufA = bufH + 2048;         // 16x128 f16
  float* v1 = sV;                       // 64x64
  float* v2 = sV + 4096;
  float* v3 = sV + 8192;
  float* v4 = sV + 12288;               // 64x128
  float* v5 = sV + 20480;

  const int hi = lane >> 4;
  const int nloc = lane & 15;

  // hoist all per-lane biases out of the time loop (LDS -> registers)
  float b_l1[4], b_r1a[4], b_r1b[4], b_r2a[8], b_r2b[8], b_r2d[8], b_tail;
#pragma unroll
  for (int nt = 0; nt < 4; ++nt) {
    b_l1[nt]  = sB[BO_L1 + nt * 16 + nloc];
    b_r1a[nt] = sB[BO_R1B1 + nt * 16 + nloc];
    b_r1b[nt] = sB[BO_R1B2 + nt * 16 + nloc];
  }
#pragma unroll
  for (int nt = 0; nt < 8; ++nt) {
    b_r2a[nt] = sB[BO_R2B1 + nt * 16 + nloc];
    b_r2b[nt] = sB[BO_R2B2 + nt * 16 + nloc];
    b_r2d[nt] = sB[BO_R2DB + nt * 16 + nloc];
  }
  b_tail = sB[BO_TAIL + nloc];

  float ysum[8];
#pragma unroll
  for (int r = 0; r < 8; ++r) ysum[r] = 0.0f;

  v8f acc[8];

  for (int t = 0; t < T_STEPS; ++t) {
    const int start = (t < LAST_START) ? t : LAST_START;

    // ===== layer1: x-window [16x201] x [201x64], LIF v1 -> spikes h in bufH
#pragma unroll
    for (int nt = 0; nt < 4; ++nt)
#pragma unroll
      for (int e = 0; e < 8; ++e) acc[nt][e] = 0.0f;
#pragma unroll
    for (int kt = 0; kt < 7; ++kt) {
      v16h a = load_a_x(xw, start, kt, lane);
#pragma unroll
      for (int nt = 0; nt < 4; ++nt) {
        v16h b = load_b_frag(sW + OFF_L1, 64, kt, nt, lane);
        acc[nt] = wmma_f16(a, b, acc[nt]);
      }
    }
#pragma unroll
    for (int nt = 0; nt < 4; ++nt) {
      const int col = nt * 16 + nloc;
#pragma unroll
      for (int r = 0; r < 8; ++r) {
        const int m = 8 * hi + r;
        float s = lif(&v1[(wave * 16 + m) * 64 + col], acc[nt][r] + b_l1[nt]);
        bufH[m * 128 + col] = (_Float16)s;
      }
    }

    // ===== res1 matmul1: LIF v2 -> spikes in bufA
    gemm_buf<2, 4>(acc, bufH, sW + OFF_R1W1, 64, lane);
#pragma unroll
    for (int nt = 0; nt < 4; ++nt) {
      const int col = nt * 16 + nloc;
#pragma unroll
      for (int r = 0; r < 8; ++r) {
        const int m = 8 * hi + r;
        float s = lif(&v2[(wave * 16 + m) * 64 + col], acc[nt][r] + b_r1a[nt]);
        bufA[m * 128 + col] = (_Float16)s;
      }
    }

    // ===== res1 matmul2: LIF v3; h = spike + h_skip -> bufH
    gemm_buf<2, 4>(acc, bufA, sW + OFF_R1W2, 64, lane);
#pragma unroll
    for (int nt = 0; nt < 4; ++nt) {
      const int col = nt * 16 + nloc;
#pragma unroll
      for (int r = 0; r < 8; ++r) {
        const int m = 8 * hi + r;
        float s = lif(&v3[(wave * 16 + m) * 64 + col], acc[nt][r] + b_r1b[nt]);
        float h = s + (float)bufH[m * 128 + col];
        bufH[m * 128 + col] = (_Float16)h;   // exact: {0,1,2}
      }
    }

    // ===== res2 matmul1: 64->128, LIF v4 -> spikes in bufA
    gemm_buf<2, 8>(acc, bufH, sW + OFF_R2W1, 128, lane);
#pragma unroll
    for (int nt = 0; nt < 8; ++nt) {
      const int col = nt * 16 + nloc;
#pragma unroll
      for (int r = 0; r < 8; ++r) {
        const int m = 8 * hi + r;
        float s = lif(&v4[(wave * 16 + m) * 128 + col], acc[nt][r] + b_r2a[nt]);
        bufA[m * 128 + col] = (_Float16)s;
      }
    }

    // ===== res2 matmul2: 128->128, LIF v5 -> spikes overwrite bufA
    gemm_buf<4, 8>(acc, bufA, sW + OFF_R2W2, 128, lane);
#pragma unroll
    for (int nt = 0; nt < 8; ++nt) {
      const int col = nt * 16 + nloc;
#pragma unroll
      for (int r = 0; r < 8; ++r) {
        const int m = 8 * hi + r;
        float s = lif(&v5[(wave * 16 + m) * 128 + col], acc[nt][r] + b_r2b[nt]);
        bufA[m * 128 + col] = (_Float16)s;
      }
    }

    // ===== res2 downsample skip: h_new = spike5 + h@dwT + db -> bufH
    gemm_buf<2, 8>(acc, bufH, sW + OFF_R2DW, 128, lane);
#pragma unroll
    for (int nt = 0; nt < 8; ++nt) {
      const int col = nt * 16 + nloc;
#pragma unroll
      for (int r = 0; r < 8; ++r) {
        const int m = 8 * hi + r;
        float hn = acc[nt][r] + b_r2d[nt] + (float)bufA[m * 128 + col];
        bufH[m * 128 + col] = (_Float16)hn;
      }
    }

    // ===== fused tail (attn + fc folded): [16x128] x [128x16], accumulate y
    gemm_buf<4, 1>(acc, bufH, sW + OFF_TAIL, 16, lane);
#pragma unroll
    for (int r = 0; r < 8; ++r) ysum[r] += acc[0][r];
  }

  // write y sum: lane holds col nloc (<10 valid), rows 8*hi + r
#pragma unroll
  for (int r = 0; r < 8; ++r) {
    if (nloc < 10) {
      int row = rowWG + wave * 16 + 8 * hi + r;
      out[row * 10 + nloc] = ysum[r] + (float)T_STEPS * b_tail;
    }
  }
}

// ======================= launch =============================================

extern "C" void kernel_launch(void* const* d_in, const int* in_sizes, int n_in,
                              void* d_out, int out_size, void* d_ws, size_t ws_size,
                              hipStream_t stream) {
  const float* x    = (const float*)d_in[0];
  const float* l1w  = (const float*)d_in[1];
  const float* l1b  = (const float*)d_in[2];
  const float* r1w1 = (const float*)d_in[3];
  const float* r1b1 = (const float*)d_in[4];
  const float* r1w2 = (const float*)d_in[5];
  const float* r1b2 = (const float*)d_in[6];
  const float* r2w1 = (const float*)d_in[7];
  const float* r2b1 = (const float*)d_in[8];
  const float* r2w2 = (const float*)d_in[9];
  const float* r2b2 = (const float*)d_in[10];
  const float* r2dw = (const float*)d_in[11];
  const float* r2db = (const float*)d_in[12];
  const float* wv   = (const float*)d_in[13];
  const float* bv   = (const float*)d_in[14];
  const float* wo   = (const float*)d_in[15];
  const float* bo   = (const float*)d_in[16];
  const float* fcw  = (const float*)d_in[17];
  const float* fcb  = (const float*)d_in[18];
  char* ws = (char*)d_ws;

  (void)hipFuncSetAttribute(reinterpret_cast<const void*>(snn_main),
                            hipFuncAttributeMaxDynamicSharedMemorySize,
                            LDS_BYTES);

  snn_prep1<<<64, 256, 0, stream>>>(l1w, l1b, r1w1, r1b1, r1w2, r1b2,
                                    r2w1, r2b1, r2w2, r2b2, r2dw, r2db,
                                    wv, bv, wo, bo, ws);
  snn_prep2<<<16, 256, 0, stream>>>(fcw, fcb, ws);
  snn_main<<<16, 128, LDS_BYTES, stream>>>(x, ws, (float*)d_out);
}
